// QKVAttention_17085379903960
// MI455X (gfx1250) — compile-verified, matrix operand order checked by reading, not run on hardware
//
#include <hip/hip_runtime.h>
#include <hip/hip_bf16.h>
#include <cstdint>

typedef _Float16 h16;
typedef _Float16 v2h  __attribute__((ext_vector_type(2)));
typedef _Float16 v4h  __attribute__((ext_vector_type(4)));
typedef _Float16 v8h  __attribute__((ext_vector_type(8)));
typedef _Float16 v16h __attribute__((ext_vector_type(16)));
typedef float    v4f  __attribute__((ext_vector_type(4)));
typedef float    v8f  __attribute__((ext_vector_type(8)));

#define SEQ_L   2048
#define CDIM    64
#define T_TILE  128
#define S_BLK   32
#define NSBLK   (SEQ_L / S_BLK)
#define QSTR    72   // f16 stride: 144B rows, 16B-aligned runs, conflict-free
#define KSTR    72
#define VSTR    40   // 80B rows
#define PSTR    40

// 1/sqrt(64) * log2(e): base-2 softmax, scale folded into Q
#define QSCALE  (0.125f * 1.44269504088896340736f)

__device__ __forceinline__ v8f wmma16(v16h a, v16h b, v8f c) {
  // D = A(16x32 f16) x B(32x16 f16) + C(16x16 f32)
  return __builtin_amdgcn_wmma_f32_16x16x32_f16(false, a, false, b, (short)0, c,
                                                false, false);
}

// Build a 16xf16 A/B operand fragment: per lane, K runs [8h,8h+8) and [16+8h,16+8h+8)
__device__ __forceinline__ v16h ld_frag(const h16* row, int h) {
  union { v16h v; v8h p[2]; } u;
  u.p[0] = *(const v8h*)(row + 8 * h);
  u.p[1] = *(const v8h*)(row + 16 + 8 * h);
  return u.v;
}

__device__ __forceinline__ v8f vzero() {
  v8f z;
#pragma unroll
  for (int j = 0; j < 8; ++j) z[j] = 0.0f;
  return z;
}

// Stage one K/V strip pair. V's s-axis is stored PERMUTED: pos(s) = s<16 ? 2s : 2(s-16)+1,
// matching the interleaved (S0,S1) P store, so P x V^T sums over the same permuted K order.
__device__ __forceinline__ void stash_kv(h16* sk, h16* sv, int cS0, int cS1, int sS,
                                         v4f ka, v4f kb, v4f va, v4f vb) {
#pragma unroll
  for (int j = 0; j < 4; ++j) {
    sk[(sS + j) * KSTR + cS0] = (h16)ka[j];
    sk[(sS + j) * KSTR + cS1] = (h16)kb[j];
  }
  const int base = (sS < 16) ? (2 * sS) : (2 * (sS - 16) + 1);  // strip stays on one side
#pragma unroll
  for (int j = 0; j < 4; ++j) {
    sv[cS0 * VSTR + base + 2 * j] = (h16)va[j];
    sv[cS1 * VSTR + base + 2 * j] = (h16)vb[j];
  }
}

__global__ __launch_bounds__(256)
void fa_qkv_kernel(const float* __restrict__ qkv, float* __restrict__ out) {
  __shared__ alignas(16) h16 sQ[T_TILE * QSTR];      // Q^T tile  [t][c]
  __shared__ alignas(16) h16 sK2[2][S_BLK * KSTR];   // K^T block [s][c], double buffered
  __shared__ alignas(16) h16 sV2[2][CDIM * VSTR];    // V  block  [c][perm(s)], double buffered
  __shared__ alignas(16) h16 sP[8][16 * PSTR];       // per-wave P scratch [t][perm(s)]

  const int tid    = threadIdx.x;
  const int wave   = tid >> 5;
  const int lane   = tid & 31;
  const int lane15 = lane & 15;
  const int hh     = lane >> 4;     // half-wave id

  const int bh    = blockIdx.x >> 4;          // head index 0..127
  const int tBase = (blockIdx.x & 15) * T_TILE;

  const float* __restrict__ qp = qkv + (size_t)(bh * 192 +   0) * SEQ_L;
  const float* __restrict__ kp = qkv + (size_t)(bh * 192 +  64) * SEQ_L;
  const float* __restrict__ vp = qkv + (size_t)(bh * 192 + 128) * SEQ_L;

  // Per-thread staging coordinates (2 strips of the 64c x 32s block)
  const int cS0 = tid >> 3;         // 0..31
  const int cS1 = cS0 + 32;         // 32..63
  const int sS  = (tid & 7) * 4;    // 0,4,..,28

  // ---- stage Q^T tile (transpose, scale by (1/8)*log2e, cvt f16) ----
  for (int i = tid; i < CDIM * T_TILE / 4; i += 256) {
    const int c = i >> 5;
    const int t = (i & 31) * 4;
    v4f q4 = *(const v4f*)(qp + (size_t)c * SEQ_L + tBase + t);
    sQ[(t + 0) * QSTR + c] = (h16)(q4.x * QSCALE);
    sQ[(t + 1) * QSTR + c] = (h16)(q4.y * QSCALE);
    sQ[(t + 2) * QSTR + c] = (h16)(q4.z * QSCALE);
    sQ[(t + 3) * QSTR + c] = (h16)(q4.w * QSCALE);
  }

  // ---- stage K/V block 0 into buffer 0 ----
  {
    v4f ka = *(const v4f*)(kp + (size_t)cS0 * SEQ_L + sS);
    v4f kb = *(const v4f*)(kp + (size_t)cS1 * SEQ_L + sS);
    v4f va = *(const v4f*)(vp + (size_t)cS0 * SEQ_L + sS);
    v4f vb = *(const v4f*)(vp + (size_t)cS1 * SEQ_L + sS);
    stash_kv(sK2[0], sV2[0], cS0, cS1, sS, ka, kb, va, vb);
  }
  __syncthreads();

  // Q operand fragments, kept in registers for the whole s loop
  const int tRow = wave * 16 + lane15;
  v16h Aq0 = ld_frag(&sQ[tRow * QSTR +  0], hh);   // c = 0..31
  v16h Aq1 = ld_frag(&sQ[tRow * QSTR + 32], hh);   // c = 32..63

  // all-ones B fragment: rowsum(P) = P x ones, lands in stats layout
  v16h Bones;
#pragma unroll
  for (int j = 0; j < 16; ++j) Bones[j] = (h16)1.0f;

  float m_i[8], l_i[8];
#pragma unroll
  for (int j = 0; j < 8; ++j) { m_i[j] = -1e30f; l_i[j] = 0.0f; }
  v8f o0 = vzero(), o1 = vzero(), o2 = vzero(), o3 = vzero();

  h16* pw = sP[wave];

  for (int sb = 0; sb < NSBLK; ++sb) {
    const int cur = sb & 1;
    const int nxt = cur ^ 1;
    const h16* sK = sK2[cur];
    const h16* sV = sV2[cur];

    // ---- issue next block's global loads early (hide HBM latency) ----
    v4f ka, kb, va, vb;
    const bool havePrefetch = (sb + 1) < NSBLK;
    if (havePrefetch) {
      const int sN = (sb + 1) * S_BLK;
      ka = *(const v4f*)(kp + (size_t)cS0 * SEQ_L + sN + sS);
      kb = *(const v4f*)(kp + (size_t)cS1 * SEQ_L + sN + sS);
      va = *(const v4f*)(vp + (size_t)cS0 * SEQ_L + sN + sS);
      vb = *(const v4f*)(vp + (size_t)cS1 * SEQ_L + sN + sS);
    }

    // ---- S = Q^T K : 16t x 32s (two 16-col subtiles, K split 2x32) ----
    v8f S0 = vzero(), S1 = vzero();
    {
      v16h B00 = ld_frag(&sK[lane15 * KSTR +  0], hh);
      v16h B01 = ld_frag(&sK[lane15 * KSTR + 32], hh);
      S0 = wmma16(Aq0, B00, S0);
      S0 = wmma16(Aq1, B01, S0);
      v16h B10 = ld_frag(&sK[(16 + lane15) * KSTR +  0], hh);
      v16h B11 = ld_frag(&sK[(16 + lane15) * KSTR + 32], hh);
      S1 = wmma16(Aq0, B10, S1);
      S1 = wmma16(Aq1, B11, S1);
    }

    // ---- row max (f16-packed butterfly over the 16-lane half) ----
    float mt[8];
#pragma unroll
    for (int j = 0; j < 8; ++j) mt[j] = fmaxf(S0[j], S1[j]);

    v2h pm[4];
#pragma unroll
    for (int i = 0; i < 4; ++i) { v2h t = { (h16)mt[2 * i], (h16)mt[2 * i + 1] }; pm[i] = t; }
#pragma unroll
    for (int off = 8; off >= 1; off >>= 1) {
#pragma unroll
      for (int i = 0; i < 4; ++i) {
        union { v2h h; int b; } u, w;
        u.h = pm[i];
        w.b = __shfl_xor(u.b, off, 32);
        pm[i] = __builtin_elementwise_max(pm[i], w.h);
      }
    }

    float alpha[8];
#pragma unroll
    for (int j = 0; j < 8; ++j) {
      const float mn = fmaxf(m_i[j], (float)pm[j >> 1][j & 1]);
      alpha[j] = __builtin_amdgcn_exp2f(m_i[j] - mn);   // base-2 softmax
      m_i[j] = mn;
    }
#pragma unroll
    for (int j = 0; j < 8; ++j) {
      S0[j] = __builtin_amdgcn_exp2f(S0[j] - m_i[j]);
      S1[j] = __builtin_amdgcn_exp2f(S1[j] - m_i[j]);
    }

    // ---- P -> per-wave LDS [t][perm(s)]: (S0,S1) pair packed per dword ----
#pragma unroll
    for (int j = 0; j < 8; ++j) {
      v2h pr = { (h16)S0[j], (h16)S1[j] };
      *(v2h*)&pw[(j + 8 * hh) * PSTR + 2 * lane15] = pr;
    }
    __builtin_amdgcn_wave_barrier();  // same-wave DS ops are in-order; block reordering

    v16h Ap = ld_frag(&pw[lane15 * PSTR], hh);  // P as A operand (16t x 32 perm-s)

    // ---- rowsum(P) via ones-matmul: element j = sum of row j+8h ----
    v8f rsv = wmma16(Ap, Bones, vzero());
#pragma unroll
    for (int j = 0; j < 8; ++j) l_i[j] = l_i[j] * alpha[j] + rsv[j];

    // ---- O = alpha*O + P V^T (4 c-chunks of 16, K in permuted s order) ----
#pragma unroll
    for (int j = 0; j < 8; ++j) {
      o0[j] *= alpha[j]; o1[j] *= alpha[j]; o2[j] *= alpha[j]; o3[j] *= alpha[j];
    }
    {
      v16h Bv0 = ld_frag(&sV[( 0 + lane15) * VSTR], hh);
      o0 = wmma16(Ap, Bv0, o0);
      v16h Bv1 = ld_frag(&sV[(16 + lane15) * VSTR], hh);
      o1 = wmma16(Ap, Bv1, o1);
      v16h Bv2 = ld_frag(&sV[(32 + lane15) * VSTR], hh);
      o2 = wmma16(Ap, Bv2, o2);
      v16h Bv3 = ld_frag(&sV[(48 + lane15) * VSTR], hh);
      o3 = wmma16(Ap, Bv3, o3);
    }

    // ---- store staged next block into the other buffer ----
    if (havePrefetch) {
      stash_kv(sK2[nxt], sV2[nxt], cS0, cS1, sS, ka, kb, va, vb);
    }
    __syncthreads();  // next buffer ready; current buffer free for overwrite
  }

  // ---- normalize and store: O[c][t], per lane 8 consecutive t per chunk ----
  float inv[8];
#pragma unroll
  for (int j = 0; j < 8; ++j) inv[j] = 1.0f / l_i[j];
  const int tOut = tBase + wave * 16 + 8 * hh;

  v8f oa[4] = { o0, o1, o2, o3 };
#pragma unroll
  for (int ch = 0; ch < 4; ++ch) {
    const int c = ch * 16 + lane15;
    float* op = out + (size_t)(bh * 64 + c) * SEQ_L + tOut;
    v4f lo = { oa[ch][0] * inv[0], oa[ch][1] * inv[1],
               oa[ch][2] * inv[2], oa[ch][3] * inv[3] };
    v4f hi = { oa[ch][4] * inv[4], oa[ch][5] * inv[5],
               oa[ch][6] * inv[6], oa[ch][7] * inv[7] };
    *(v4f*)op       = lo;
    *(v4f*)(op + 4) = hi;
  }
}

extern "C" void kernel_launch(void* const* d_in, const int* in_sizes, int n_in,
                              void* d_out, int out_size, void* d_ws, size_t ws_size,
                              hipStream_t stream) {
  (void)in_sizes; (void)n_in; (void)out_size; (void)d_ws; (void)ws_size;
  const float* qkv = (const float*)d_in[0];
  float* out = (float*)d_out;
  const int grid = 128 * (SEQ_L / T_TILE);   // 2048 workgroups
  fa_qkv_kernel<<<grid, 256, 0, stream>>>(qkv, out);
}
